// LanguageModel_7859790151898
// MI455X (gfx1250) — matrix operand order repaired and checked by reading, not compiled, not run on hardware
//
#include <hip/hip_runtime.h>
#include <hip/hip_bf16.h>
#include <stdint.h>

// ---------------------------------------------------------------------------
// GPT-mini forward for MI455X (gfx1250, wave32, WMMA + async Global->LDS).
//   x = tok_emb[idx] + pos_emb            (f32 -> f16)
//   qkv = x @ [Wq|Wk|Wv]                  (WMMA, async-LDS staged B)
//   att = flash-attention(q,k,v)          (WMMA + online softmax)
//   h   = relu(att @ Wf + bf)             (WMMA)
//   logits = h @ Wlm                      (WMMA, f32 out -> d_out)
//   loss = mean CE(logits, targets)       (block reductions)
// ---------------------------------------------------------------------------

#define VOCAB 32000
#define EMBED 1024
#define TSEQ  2048
#define NB    2
#define HEADS 16
#define HDIM  64
#define MROWS (NB*TSEQ)   // 4096

typedef __attribute__((ext_vector_type(16))) _Float16     v16h;
typedef __attribute__((ext_vector_type(8)))  float        v8f;
typedef __attribute__((ext_vector_type(8)))  unsigned int v8u;

union Frag { unsigned int u[8]; v8u v; };

__device__ __forceinline__ v16h as_v16h(const Frag& f) {
  return __builtin_bit_cast(v16h, f.v);
}

// A-fragment (16x32 f16) K offset for VGPR j, lane-half g  (ISA §7.12.2):
// lanes 0-15 hold K=0..7 then 16..23; lanes 16-31 hold K=8..15 then 24..31,
// packed as (K, K+1) pairs per 32-bit VGPR.
__device__ __forceinline__ int a_koff(int j, int g) {
  return (j < 4) ? (8*g + 2*j) : (16 + 8*g + 2*(j-4));
}

// ---------------------------------------------------------------------------
// CDNA5 async Global->LDS copy engine (CDNA5 ISA).
// One instruction per wave: each lane copies 16 bytes. VDST VGPR carries the
// per-lane LDS byte address; tracked by ASYNCcnt.
// ---------------------------------------------------------------------------
__device__ __forceinline__ void async_copy_b128(unsigned int lds_off,
                                                unsigned long long gaddr) {
  asm volatile("global_load_async_to_lds_b128 %0, %1, off"
               :: "v"(lds_off), "v"(gaddr) : "memory");
}
__device__ __forceinline__ void wait_asynccnt0() {
  asm volatile("s_wait_asynccnt 0" ::: "memory");
}

// ---------------------------------------------------------------------------
// Weight conversion: W[K][N] f32 -> WT[N][K] f16 (column-major B for WMMA so
// the contraction dim is contiguous -> 128-bit fragment loads).
// ---------------------------------------------------------------------------
__global__ void k_cvt_T(const float* __restrict__ W, _Float16* __restrict__ WT,
                        int K, int N) {
  size_t i = (size_t)blockIdx.x * blockDim.x + threadIdx.x;
  if (i >= (size_t)K * N) return;
  int n = (int)(i / K), k = (int)(i % K);
  WT[i] = (_Float16)W[(size_t)k * N + n];
}

__global__ void k_cvt_qkv(const float* __restrict__ Wq, const float* __restrict__ Wk,
                          const float* __restrict__ Wv, _Float16* __restrict__ WT) {
  size_t i = (size_t)blockIdx.x * blockDim.x + threadIdx.x;  // over 3*E*E
  if (i >= (size_t)3 * EMBED * EMBED) return;
  int n = (int)(i / EMBED), k = (int)(i % EMBED);
  int sel = n >> 10, c = n & 1023;                 // output col within Wq/Wk/Wv
  const float* src = (sel == 0) ? Wq : (sel == 1) ? Wk : Wv;
  WT[i] = (_Float16)src[(size_t)k * EMBED + c];    // [E, H*D] flattened
}

// ---------------------------------------------------------------------------
// Embedding: X[row][e] = tok_emb[idx[row]][e] + pos_emb[row % T][e]  (f16)
// ---------------------------------------------------------------------------
__global__ void k_embed(const int* __restrict__ idx, const float* __restrict__ tok,
                        const float* __restrict__ pos, _Float16* __restrict__ X) {
  size_t i = (size_t)blockIdx.x * blockDim.x + threadIdx.x;  // MROWS*EMBED
  int row = (int)(i >> 10), e = (int)(i & 1023);
  int t = row & (TSEQ - 1);
  int token = idx[row];
  X[i] = (_Float16)(tok[(size_t)token * EMBED + e] + pos[(size_t)t * EMBED + e]);
}

// ---------------------------------------------------------------------------
// WMMA GEMM: C[M][N] = A[M][K] @ BT[N][K]^T  (+bias, relu, f32/f16 out).
// 4 waves / block, each owning a 64-row M band of a 256x64 C block; the 64x32
// B tile per K-step is staged into LDS by the async engine (double-buffered,
// s_wait_asynccnt + barrier) and shared by all 4 waves -> 4x less L2 traffic
// on the weight stream (dominant for the 32000-col LM head).
// ---------------------------------------------------------------------------
template<bool RELU, bool OUTF32>
__global__ void k_gemm(const _Float16* __restrict__ A, const _Float16* __restrict__ BT,
                       const float* __restrict__ bias, void* __restrict__ Cout,
                       int M, int N, int K) {
  const int tid  = threadIdx.x;
  const int wave = tid >> 5, lane = tid & 31;
  const int g = lane >> 4, nn = lane & 15;
  const int m0 = blockIdx.y * 256 + wave * 64;
  const int n0 = blockIdx.x * 64;
  (void)M;

  __shared__ __align__(16) _Float16 Bsm[2][64 * 32];

  // Cooperative async stage of the 64x32 B tile for K-step k0 into Bsm[buf].
  // 256 16-byte chunks; 128 threads x 2 -> 8 wave-instructions per block.
  auto stage = [&](int buf, int k0) {
    const unsigned int base = (unsigned int)(uintptr_t)&Bsm[buf][0];
#pragma unroll
    for (int i = 0; i < 2; ++i) {
      const int c = tid + 128 * i;
      const int row = c >> 2, col = (c & 3) * 8;     // 4 chunks per 32-elem row
      const unsigned long long gaddr =
          (unsigned long long)(uintptr_t)(BT + (size_t)(n0 + row) * K + k0 + col);
      async_copy_b128(base + (unsigned int)c * 16, gaddr);
    }
  };

  const v8f vzero = {0.f,0.f,0.f,0.f,0.f,0.f,0.f,0.f};
  v8f acc[4][4];
  for (int mi = 0; mi < 4; ++mi)
    for (int ni = 0; ni < 4; ++ni) acc[mi][ni] = vzero;

  stage(0, 0);
  int buf = 0;
  for (int k0 = 0; k0 < K; k0 += 32, buf ^= 1) {
    wait_asynccnt0();        // own async copies for Bsm[buf] complete
    __syncthreads();         // everyone's copies visible; prev reads finished
    if (k0 + 32 < K) stage(buf ^ 1, k0 + 32);

    Frag a[4], b[4];
#pragma unroll
    for (int mi = 0; mi < 4; ++mi) {
      const _Float16* ar = A + (size_t)(m0 + 16*mi + nn) * K + k0;
      __builtin_prefetch(ar + 32, 0, 1);             // global_prefetch_b8
#pragma unroll
      for (int j = 0; j < 8; ++j)
        a[mi].u[j] = *(const unsigned int*)(ar + 16*g + 2*j);
    }
#pragma unroll
    for (int ni = 0; ni < 4; ++ni) {
      const _Float16* br = &Bsm[buf][(16*ni + nn) * 32 + 16*g];
#pragma unroll
      for (int j = 0; j < 8; ++j)
        b[ni].u[j] = *(const unsigned int*)(br + 2*j);  // ds_load_b128 pairs
    }
#pragma unroll
    for (int mi = 0; mi < 4; ++mi) {
      v16h av = as_v16h(a[mi]);
#pragma unroll
      for (int ni = 0; ni < 4; ++ni)
        acc[mi][ni] = __builtin_amdgcn_wmma_f32_16x16x32_f16(
            false, av, false, as_v16h(b[ni]), (short)0, acc[mi][ni], false, false);
    }
  }

  // Epilogue: C/D layout — VGPR r holds row m = r + 8*g, col n = lane&15.
#pragma unroll
  for (int mi = 0; mi < 4; ++mi)
#pragma unroll
    for (int ni = 0; ni < 4; ++ni)
#pragma unroll
      for (int r = 0; r < 8; ++r) {
        int m = m0 + 16*mi + r + 8*g;
        int n = n0 + 16*ni + nn;
        float v = acc[mi][ni][r];
        if (bias) v += bias[n];
        if (RELU) v = v > 0.f ? v : 0.f;
        if (OUTF32) ((float*)Cout)[(size_t)m * N + n] = v;
        else        ((_Float16*)Cout)[(size_t)m * N + n] = (_Float16)v;
      }
}

// ---------------------------------------------------------------------------
// Flash attention: one wave per (b, h, 16-query tile). Keys processed 32 at a
// time: QK^T via 4 WMMAs, online softmax (shfl_xor row reductions within each
// 16-lane half), P reshaped C->A layout through 1KB LDS, P@V via 4 WMMAs.
// qkv layout: [b*T + t][3E], cols 0..E-1 = q (h-major), E.. = k, 2E.. = v.
// ---------------------------------------------------------------------------
__global__ void k_attn(const _Float16* __restrict__ qkv, _Float16* __restrict__ att) {
  const int lane = threadIdx.x, g = lane >> 4, nn = lane & 15;
  const int wid = blockIdx.x;
  const int qt = wid & 127;
  const int h  = (wid >> 7) & 15;
  const int b  = wid >> 11;
  const int qbase = qt * 16;
  const size_t rs = 3 * EMBED;
  const int hoff = h * HDIM;
  const float scale = 0.03125f;                    // EMBED**-0.5

  // Q fragments: 16x64 tile as two 16x32 A-frags along the contraction dim.
  Frag qa[2];
  {
    const _Float16* qr = qkv + (size_t)(b*TSEQ + qbase + nn) * rs + hoff;
#pragma unroll
    for (int ks = 0; ks < 2; ++ks)
#pragma unroll
      for (int j = 0; j < 8; ++j)
        qa[ks].u[j] = *(const unsigned int*)(qr + 32*ks + 16*g + 2*j);
  }

  const v8f vzero = {0.f,0.f,0.f,0.f,0.f,0.f,0.f,0.f};
  v8f acc[4];                                      // out 16x64 = 4 N-subtiles
  for (int ni = 0; ni < 4; ++ni) acc[ni] = vzero;
  float Mrow[8], Lrow[8];
#pragma unroll
  for (int r = 0; r < 8; ++r) { Mrow[r] = -3.0e38f; Lrow[r] = 0.f; }

  __shared__ __align__(16) _Float16 Plds[16 * 32];

  const int kb_end = (qbase + 15) >> 5;
  for (int kb = 0; kb <= kb_end; ++kb) {
    // ---- scores: two 16x16 tiles covering 32 key positions ----
    v8f s[2];
#pragma unroll
    for (int kn = 0; kn < 2; ++kn) {
      const int kp = kb*32 + 16*kn + nn;           // B-frag lane = key position
      const _Float16* kr = qkv + (size_t)(b*TSEQ + kp) * rs + EMBED + hoff;
      Frag kf0, kf1;
#pragma unroll
      for (int j = 0; j < 8; ++j) {
        kf0.u[j] = *(const unsigned int*)(kr      + 16*g + 2*j);
        kf1.u[j] = *(const unsigned int*)(kr + 32 + 16*g + 2*j);
      }
      v8f z = vzero;
      z = __builtin_amdgcn_wmma_f32_16x16x32_f16(false, as_v16h(qa[0]), false,
                                                 as_v16h(kf0), (short)0, z, false, false);
      z = __builtin_amdgcn_wmma_f32_16x16x32_f16(false, as_v16h(qa[1]), false,
                                                 as_v16h(kf1), (short)0, z, false, false);
      s[kn] = z;
    }
    // ---- scale + causal mask + running row max ----
    float rm[8];
#pragma unroll
    for (int r = 0; r < 8; ++r) {
      const int qp = qbase + r + 8*g;
      float v0 = s[0][r] * scale, v1 = s[1][r] * scale;
      if (kb*32 + nn      > qp) v0 = -3.0e38f;
      if (kb*32 + 16 + nn > qp) v1 = -3.0e38f;
      s[0][r] = v0; s[1][r] = v1;
      rm[r] = fmaxf(v0, v1);
    }
#pragma unroll
    for (int off = 1; off < 16; off <<= 1)
#pragma unroll
      for (int r = 0; r < 8; ++r) rm[r] = fmaxf(rm[r], __shfl_xor(rm[r], off, 32));

    float f[8];
#pragma unroll
    for (int r = 0; r < 8; ++r) {
      float nm = fmaxf(Mrow[r], rm[r]);
      f[r] = __expf(Mrow[r] - nm);
      Mrow[r] = nm;
    }
    float sm[8];
#pragma unroll
    for (int r = 0; r < 8; ++r) {
      float p0 = __expf(s[0][r] - Mrow[r]);
      float p1 = __expf(s[1][r] - Mrow[r]);
      s[0][r] = p0; s[1][r] = p1; sm[r] = p0 + p1;
    }
#pragma unroll
    for (int off = 1; off < 16; off <<= 1)
#pragma unroll
      for (int r = 0; r < 8; ++r) sm[r] += __shfl_xor(sm[r], off, 32);
#pragma unroll
    for (int r = 0; r < 8; ++r) Lrow[r] = Lrow[r] * f[r] + sm[r];
#pragma unroll
    for (int ni = 0; ni < 4; ++ni)
#pragma unroll
      for (int r = 0; r < 8; ++r) acc[ni][r] *= f[r];

    // ---- P: C-layout -> A-layout via LDS ----
    __syncthreads();
#pragma unroll
    for (int kn = 0; kn < 2; ++kn)
#pragma unroll
      for (int r = 0; r < 8; ++r)
        Plds[(r + 8*g)*32 + 16*kn + nn] = (_Float16)s[kn][r];
    __syncthreads();
    Frag pa;
#pragma unroll
    for (int j = 0; j < 8; ++j)
      pa.u[j] = *(const unsigned int*)(&Plds[nn*32 + 16*g + 2*j]);

    // ---- out += P @ V (V rows are key positions; pack pairs manually) ----
#pragma unroll
    for (int ni = 0; ni < 4; ++ni) {
      const int d = 16*ni + nn;
      Frag vb;
#pragma unroll
      for (int j = 0; j < 8; ++j) {
        const int k1 = 16*g + 2*j;
        const size_t base = (size_t)(b*TSEQ + kb*32 + k1) * rs + 2*EMBED + hoff + d;
        unsigned int lo = __builtin_bit_cast(unsigned short, qkv[base]);
        unsigned int hi = __builtin_bit_cast(unsigned short, qkv[base + rs]);
        vb.u[j] = lo | (hi << 16);
      }
      acc[ni] = __builtin_amdgcn_wmma_f32_16x16x32_f16(false, as_v16h(pa), false,
                                                       as_v16h(vb), (short)0, acc[ni], false, false);
    }
  }

  // normalize + write concat-heads output [row][h*64+d] (f16)
#pragma unroll
  for (int ni = 0; ni < 4; ++ni)
#pragma unroll
    for (int r = 0; r < 8; ++r) {
      const int m = qbase + r + 8*g;
      att[(size_t)(b*TSEQ + m) * EMBED + hoff + 16*ni + nn] =
          (_Float16)(acc[ni][r] / Lrow[r]);
    }
}

// ---------------------------------------------------------------------------
// Cross-entropy: one block per row; max + sum-exp reductions over VOCAB.
// ---------------------------------------------------------------------------
__global__ void k_loss_init(float* loss) { *loss = 0.f; }

__global__ void k_loss(const float* __restrict__ logits, const int* __restrict__ targets,
                       float* __restrict__ loss) {
  const int row = blockIdx.x, tid = threadIdx.x;
  const float* lr = logits + (size_t)row * VOCAB;
  __shared__ float red[256];
  float mx = -3.0e38f;
  for (int i = tid; i < VOCAB; i += 256) mx = fmaxf(mx, lr[i]);
  red[tid] = mx; __syncthreads();
  for (int s = 128; s > 0; s >>= 1) {
    if (tid < s) red[tid] = fmaxf(red[tid], red[tid + s]);
    __syncthreads();
  }
  mx = red[0]; __syncthreads();
  float sum = 0.f;
  for (int i = tid; i < VOCAB; i += 256) sum += __expf(lr[i] - mx);
  red[tid] = sum; __syncthreads();
  for (int s = 128; s > 0; s >>= 1) {
    if (tid < s) red[tid] += red[tid + s];
    __syncthreads();
  }
  if (tid == 0) {
    float lse = mx + __logf(red[0]);
    atomicAdd(loss, (lse - lr[targets[row]]) * (1.0f / (float)MROWS));
  }
}

// ---------------------------------------------------------------------------
extern "C" void kernel_launch(void* const* d_in, const int* in_sizes, int n_in,
                              void* d_out, int out_size, void* d_ws, size_t ws_size,
                              hipStream_t stream) {
  (void)in_sizes; (void)n_in; (void)out_size; (void)ws_size;
  const int*   idx     = (const int*)d_in[0];
  const int*   targets = (const int*)d_in[1];
  const float* tok_emb = (const float*)d_in[2];
  const float* pos_emb = (const float*)d_in[3];
  const float* Wq      = (const float*)d_in[4];
  const float* Wk      = (const float*)d_in[5];
  const float* Wv      = (const float*)d_in[6];
  const float* Wf      = (const float*)d_in[7];
  const float* bf      = (const float*)d_in[8];
  const float* Wlm     = (const float*)d_in[9];
  float* out = (float*)d_out;

  char* ws = (char*)d_ws;
  size_t off = 0;
  auto alloc = [&](size_t bytes) -> void* {
    void* p = ws + off;
    off += (bytes + 255) & ~(size_t)255;
    return p;
  };
  _Float16* X     = (_Float16*)alloc((size_t)MROWS * EMBED * 2);       // 8 MB
  _Float16* WQKVT = (_Float16*)alloc((size_t)3 * EMBED * EMBED * 2);   // 6 MB
  _Float16* QKV   = (_Float16*)alloc((size_t)MROWS * 3 * EMBED * 2);   // 24 MB
  _Float16* ATT   = (_Float16*)alloc((size_t)MROWS * EMBED * 2);       // 8 MB
  _Float16* WFT   = (_Float16*)alloc((size_t)EMBED * EMBED * 2);       // 2 MB
  _Float16* H     = (_Float16*)alloc((size_t)MROWS * EMBED * 2);       // 8 MB
  _Float16* WLMT  = (_Float16*)alloc((size_t)VOCAB * EMBED * 2);       // 64 MB

  // weight conversion (f32 -> f16, transposed so contraction dim contiguous)
  k_cvt_qkv<<<(3*EMBED*EMBED)/256, 256, 0, stream>>>(Wq, Wk, Wv, WQKVT);
  k_cvt_T<<<(EMBED*EMBED)/256, 256, 0, stream>>>(Wf, WFT, EMBED, EMBED);
  k_cvt_T<<<((size_t)VOCAB*EMBED)/256, 256, 0, stream>>>(Wlm, WLMT, EMBED, VOCAB);

  // embedding
  k_embed<<<(MROWS*EMBED)/256, 256, 0, stream>>>(idx, tok_emb, pos_emb, X);

  // qkv projection: [4096,1024] @ [1024,3072]
  k_gemm<false, false><<<dim3(3*EMBED/64, MROWS/256), 128, 0, stream>>>(
      X, WQKVT, nullptr, QKV, MROWS, 3*EMBED, EMBED);

  // flash attention: B*H*(T/16) waves
  k_attn<<<NB*HEADS*(TSEQ/16), 32, 0, stream>>>(QKV, ATT);

  // FFN: relu([4096,1024] @ [1024,1024] + bf)
  k_gemm<true, false><<<dim3(EMBED/64, MROWS/256), 128, 0, stream>>>(
      ATT, WFT, bf, H, MROWS, EMBED, EMBED);

  // LM head: [4096,1024] @ [1024,32000] -> f32 logits straight into d_out
  k_gemm<false, true><<<dim3(VOCAB/64, MROWS/256), 128, 0, stream>>>(
      H, WLMT, nullptr, out, MROWS, VOCAB, EMBED);

  // cross-entropy loss -> d_out[MROWS*VOCAB]
  float* loss = out + (size_t)MROWS * VOCAB;
  k_loss_init<<<1, 1, 0, stream>>>(loss);
  k_loss<<<MROWS, 256, 0, stream>>>(out, targets, loss);
}
